// QuantumAttention_65481071408115
// MI455X (gfx1250) — compile-verified, matrix operand order checked by reading, not run on hardware
//
#include <hip/hip_runtime.h>
#include <hip/hip_bf16.h>
#include <stdint.h>

typedef __attribute__((ext_vector_type(16))) _Float16 v16h;
typedef __attribute__((ext_vector_type(4)))  _Float16 v4h;
typedef __attribute__((ext_vector_type(8)))  float    v8f;
typedef __attribute__((ext_vector_type(4)))  int      v4i_t;

union Frag {
  uint4 u[2];   // 32 bytes = 16 f16
  v16h  h;
};

#define WMMA_F16(A, B, C) \
  __builtin_amdgcn_wmma_f32_16x16x32_f16(false, (A), false, (B), (short)0, (C), false, false)

static constexpr int E_DIM = 1024;
static constexpr int HEADS = 16;
static constexpr int DK    = 64;
static constexpr int SEQ   = 2048;
static constexpr int BATCH = 2;
static constexpr int MTOT  = BATCH * SEQ;   // 4096
static constexpr int LDS_LD = 40;           // padded row stride (f16) for LDS tiles

// ---------------------------------------------------------------------------
// CDNA5 async global->LDS path (guarded: falls back to sync staging)
// Builtin signature (from compiler diagnostic): (v4i as1*, v4i as3*, Ii, Ii)
// ---------------------------------------------------------------------------
#if __has_builtin(__builtin_amdgcn_global_load_async_to_lds_b128)
#define HAVE_ASYNC_LDS 1
typedef __attribute__((address_space(1))) v4i_t* gv4i;
typedef __attribute__((address_space(3))) v4i_t* lv4i;
__device__ __forceinline__ void async_b128(const _Float16* g, _Float16* l) {
  __builtin_amdgcn_global_load_async_to_lds_b128((gv4i)(void*)g, (lv4i)(void*)l, 0, 0);
}
__device__ __forceinline__ void wait_async0() {
#if __has_builtin(__builtin_amdgcn_s_wait_asynccnt)
  __builtin_amdgcn_s_wait_asynccnt(0);
#else
  asm volatile("s_wait_asynccnt 0x0" ::: "memory");
#endif
}
#else
#define HAVE_ASYNC_LDS 0
__device__ __forceinline__ void wait_async0() {}
#endif

__device__ __forceinline__ void stage16(const _Float16* g, _Float16* l) {
#if HAVE_ASYNC_LDS
  async_b128(g, l);
  async_b128(g + 8, l + 8);
#else
  uint4 t0 = *(const uint4*)g;
  uint4 t1 = *(const uint4*)(g + 8);
  *(uint4*)l       = t0;
  *(uint4*)(l + 8) = t1;
#endif
}

__device__ __forceinline__ Frag lds_frag(const _Float16* base) {
  Frag f;
  f.u[0] = *(const uint4*)base;
  f.u[1] = *(const uint4*)(base + 16);
  return f;
}

// ---------------------------------------------------------------------------
// f32 -> f16 cast (vectorized; all sizes are multiples of 4)
// ---------------------------------------------------------------------------
__global__ void cast_f32_to_f16(const float* __restrict__ src,
                                _Float16* __restrict__ dst, int n4) {
  int i = blockIdx.x * blockDim.x + threadIdx.x;
  int stride = gridDim.x * blockDim.x;
  for (; i < n4; i += stride) {
    float4 f = ((const float4*)src)[i];
    v4h h;
    h.x = (_Float16)f.x; h.y = (_Float16)f.y;
    h.z = (_Float16)f.z; h.w = (_Float16)f.w;
    ((v4h*)dst)[i] = h;
  }
}

// ---------------------------------------------------------------------------
// Shared 128x128 GEMM core: acc(64x32 per wave) += A[128xE] @ W[128xE]^T slab.
// Double-buffered LDS K-slabs of 32, staged with async loads when available.
// ---------------------------------------------------------------------------
__device__ __forceinline__ void gemm_core(const _Float16* __restrict__ A,
                                          const _Float16* __restrict__ W,
                                          _Float16* As, _Float16* Bs,
                                          int mBlock, int nBlock,
                                          v8f acc[4][2]) {
  const int tid   = threadIdx.x;
  const int lane  = tid & 31;
  const int wave  = tid >> 5;
  const int waveM = wave >> 2;     // 0..1
  const int waveN = wave & 3;      // 0..3
  const int l15   = lane & 15;
  const int half  = lane >> 4;
  const int r  = tid >> 1;         // staging row 0..127
  const int cc = (tid & 1) * 16;   // staging col 0 or 16

  auto stage = [&](int k0, int buf) {
    const _Float16* ga = A + (size_t)(mBlock + r) * E_DIM + k0 + cc;
    const _Float16* gb = W + (size_t)(nBlock + r) * E_DIM + k0 + cc;
    _Float16* la = As + buf * (128 * LDS_LD) + r * LDS_LD + cc;
    _Float16* lb = Bs + buf * (128 * LDS_LD) + r * LDS_LD + cc;
    stage16(ga, la);
    stage16(gb, lb);
  };

  stage(0, 0);
  int buf = 0;
  for (int k0 = 0; k0 < E_DIM; k0 += 32) {
    wait_async0();
    __syncthreads();
    if (k0 + 32 < E_DIM) stage(k0 + 32, buf ^ 1);

    const _Float16* Ab = As + buf * (128 * LDS_LD);
    const _Float16* Bb = Bs + buf * (128 * LDS_LD);
    Frag a[4], b[2];
#pragma unroll
    for (int mt = 0; mt < 4; ++mt)
      a[mt] = lds_frag(Ab + (waveM * 64 + mt * 16 + l15) * LDS_LD + half * 8);
#pragma unroll
    for (int nt = 0; nt < 2; ++nt)
      b[nt] = lds_frag(Bb + (waveN * 32 + nt * 16 + l15) * LDS_LD + half * 8);
#pragma unroll
    for (int mt = 0; mt < 4; ++mt)
#pragma unroll
      for (int nt = 0; nt < 2; ++nt)
        acc[mt][nt] = WMMA_F16(a[mt].h, b[nt].h, acc[mt][nt]);
    buf ^= 1;
  }
}

// ---------------------------------------------------------------------------
// Fused QKV projection: y = x @ W.T.
// Q,K stored head-major [B,H,S,dk]; V stored d-major [B,H,dk,S] so the
// attention P@V B-fragments are contiguous global loads (no LDS transpose).
// ---------------------------------------------------------------------------
__global__ void __launch_bounds__(256)
qkv_gemm(const _Float16* __restrict__ X,
         const _Float16* __restrict__ Wq,
         const _Float16* __restrict__ Wk,
         const _Float16* __restrict__ Wv,
         _Float16* __restrict__ Qo,
         _Float16* __restrict__ Ko,
         _Float16* __restrict__ Vo) {
  __shared__ __align__(16) _Float16 As[2 * 128 * LDS_LD];
  __shared__ __align__(16) _Float16 Bs[2 * 128 * LDS_LD];

  const int lane  = threadIdx.x & 31;
  const int wave  = threadIdx.x >> 5;
  const int waveM = wave >> 2;
  const int waveN = wave & 3;
  const int l15   = lane & 15;
  const int half  = lane >> 4;
  const int mBlock = blockIdx.y * 128;
  const int nBlock = blockIdx.x * 128;

  const _Float16* W = (blockIdx.z == 0) ? Wq : (blockIdx.z == 1) ? Wk : Wv;
  _Float16* Dst     = (blockIdx.z == 0) ? Qo : (blockIdx.z == 1) ? Ko : Vo;

  v8f zero = {0.f, 0.f, 0.f, 0.f, 0.f, 0.f, 0.f, 0.f};
  v8f acc[4][2];
#pragma unroll
  for (int mt = 0; mt < 4; ++mt)
#pragma unroll
    for (int nt = 0; nt < 2; ++nt) acc[mt][nt] = zero;

  gemm_core(X, W, As, Bs, mBlock, nBlock, acc);

  const bool vT = (blockIdx.z == 2);
#pragma unroll
  for (int mt = 0; mt < 4; ++mt)
#pragma unroll
    for (int nt = 0; nt < 2; ++nt)
#pragma unroll
      for (int j = 0; j < 8; ++j) {
        int m  = mBlock + waveM * 64 + mt * 16 + j + 8 * half;
        int n  = nBlock + waveN * 32 + nt * 16 + l15;
        int b_ = m >> 11, s_ = m & (SEQ - 1);
        int h_ = n >> 6,  d_ = n & (DK - 1);
        _Float16 v = (_Float16)acc[mt][nt][j];
        if (vT)
          Dst[((size_t)(b_ * HEADS + h_) * DK + d_) * SEQ + s_] = v;   // [B,H,dk,S]
        else
          Dst[((size_t)(b_ * HEADS + h_) * SEQ + s_) * DK + d_] = v;   // [B,H,S,dk]
      }
}

// ---------------------------------------------------------------------------
// Flash attention: block = (b,h) x 64 query rows; 4 waves x 16 rows; no
// workgroup barriers (P tile lives in per-wave LDS; same-wave DS is in-order).
// ---------------------------------------------------------------------------
__global__ void __launch_bounds__(128)
attn_kernel(const _Float16* __restrict__ Qg,
            const _Float16* __restrict__ Kg,
            const _Float16* __restrict__ Vg,   // d-major [B,H,dk,S]
            const float* __restrict__ qparams,
            _Float16* __restrict__ Ctx) {
  __shared__ __align__(16) _Float16 Pl[4 * 16 * 72];

  const int lane = threadIdx.x & 31;
  const int wave = threadIdx.x >> 5;
  const int l15  = lane & 15;
  const int half = lane >> 4;

  const int bh    = blockIdx.y;   // b*16+h
  const int qTile = blockIdx.x;
  const size_t base = (size_t)bh * SEQ * DK;
  const _Float16* Qp = Qg + base;
  const _Float16* Kp = Kg + base;
  const _Float16* Vp = Vg + base;          // rows are dk, cols are keys
  const int qRow0 = qTile * 64 + wave * 16;

  const float cq = __cosf(qparams[0] + qparams[1]);   // RX circuit collapses

  Frag aQ[2];
#pragma unroll
  for (int ks = 0; ks < 2; ++ks) {
    const _Float16* p = Qp + (size_t)(qRow0 + l15) * DK + ks * 32 + half * 8;
    aQ[ks].u[0] = *(const uint4*)p;
    aQ[ks].u[1] = *(const uint4*)(p + 16);
  }

  v8f zero = {0.f, 0.f, 0.f, 0.f, 0.f, 0.f, 0.f, 0.f};
  v8f o[4];
#pragma unroll
  for (int dt = 0; dt < 4; ++dt) o[dt] = zero;
  float mrow[8], lrow[8];
#pragma unroll
  for (int j = 0; j < 8; ++j) { mrow[j] = -INFINITY; lrow[j] = 0.f; }

  _Float16* Pw = Pl + wave * 16 * 72;

  for (int kc = 0; kc < SEQ / 64; ++kc) {
    // ---- scores S = Q K^T (16 x 64): B-fragments contiguous from K ----
    v8f s[4];
#pragma unroll
    for (int nt = 0; nt < 4; ++nt) s[nt] = zero;
#pragma unroll
    for (int ks = 0; ks < 2; ++ks)
#pragma unroll
      for (int nt = 0; nt < 4; ++nt) {
        Frag bK;
        const _Float16* p =
            Kp + (size_t)(kc * 64 + nt * 16 + l15) * DK + ks * 32 + half * 8;
        bK.u[0] = *(const uint4*)p;
        bK.u[1] = *(const uint4*)(p + 16);
        s[nt] = WMMA_F16(aQ[ks].h, bK.h, s[nt]);
      }
    if (kc + 1 < SEQ / 64) {
      __builtin_prefetch(Kp + (size_t)((kc + 1) * 64 + l15) * DK, 0, 1);
      __builtin_prefetch(Vp + (size_t)l15 * SEQ + (kc + 1) * 64, 0, 1);
    }

    // ---- online softmax; stat register j <-> row j + 8*laneHalf ----
    float cmax[8];
#pragma unroll
    for (int j = 0; j < 8; ++j) {
      float mx = -INFINITY;
#pragma unroll
      for (int nt = 0; nt < 4; ++nt) {
        float v = s[nt][j] * 0.125f;   // 1/sqrt(dk)
        s[nt][j] = v;
        mx = fmaxf(mx, v);
      }
      cmax[j] = mx;
    }
#pragma unroll
    for (int msk = 1; msk <= 8; msk <<= 1)
#pragma unroll
      for (int j = 0; j < 8; ++j) cmax[j] = fmaxf(cmax[j], __shfl_xor(cmax[j], msk));

    float alpha[8], csum[8];
#pragma unroll
    for (int j = 0; j < 8; ++j) {
      float mnew = fmaxf(mrow[j], cmax[j]);
      alpha[j] = __expf(mrow[j] - mnew);
      mrow[j] = mnew;
    }
#pragma unroll
    for (int j = 0; j < 8; ++j) {
      float sum = 0.f;
#pragma unroll
      for (int nt = 0; nt < 4; ++nt) {
        float p = __expf(s[nt][j] - mrow[j]);
        sum += p;
        Pw[(j + 8 * half) * 72 + nt * 16 + l15] = (_Float16)p;
      }
      csum[j] = sum;
    }
#pragma unroll
    for (int msk = 1; msk <= 8; msk <<= 1)
#pragma unroll
      for (int j = 0; j < 8; ++j) csum[j] += __shfl_xor(csum[j], msk);
#pragma unroll
    for (int j = 0; j < 8; ++j) lrow[j] = lrow[j] * alpha[j] + csum[j];
#pragma unroll
    for (int dt = 0; dt < 4; ++dt)
#pragma unroll
      for (int j = 0; j < 8; ++j) o[dt][j] *= alpha[j];

    // ---- O += P @ V : A from per-wave LDS, B contiguous from V^T global ----
#pragma unroll
    for (int ks = 0; ks < 2; ++ks) {
      Frag aP;
      const _Float16* p = Pw + l15 * 72 + ks * 32 + half * 8;
      aP.u[0] = *(const uint4*)p;
      aP.u[1] = *(const uint4*)(p + 16);
#pragma unroll
      for (int dt = 0; dt < 4; ++dt) {
        Frag bV;
        const _Float16* q =
            Vp + (size_t)(dt * 16 + l15) * SEQ + kc * 64 + ks * 32 + half * 8;
        bV.u[0] = *(const uint4*)q;
        bV.u[1] = *(const uint4*)(q + 16);
        o[dt] = WMMA_F16(aP.h, bV.h, o[dt]);
      }
    }
  }

  // ---- epilogue: normalize, add quantum constant on d<4, store ctx ----
  const int b_ = bh >> 4;
  const int hcol = (bh & 15) * DK;
#pragma unroll
  for (int j = 0; j < 8; ++j) {
    float inv = 1.0f / lrow[j];
    int srow = qRow0 + j + 8 * half;
    size_t ob = ((size_t)(b_ * SEQ + srow)) * E_DIM + hcol;
#pragma unroll
    for (int dt = 0; dt < 4; ++dt) {
      float v = o[dt][j] * inv;
      if (dt == 0 && l15 < 4) v += cq;   // N_QUBITS = 4
      Ctx[ob + dt * 16 + l15] = (_Float16)v;
    }
  }
}

// ---------------------------------------------------------------------------
// Output projection: out = ctx @ Wo.T, f32 row-major store.
// ---------------------------------------------------------------------------
__global__ void __launch_bounds__(256)
out_gemm(const _Float16* __restrict__ Ctx,
         const _Float16* __restrict__ Wo,
         float* __restrict__ Out) {
  __shared__ __align__(16) _Float16 As[2 * 128 * LDS_LD];
  __shared__ __align__(16) _Float16 Bs[2 * 128 * LDS_LD];

  const int lane  = threadIdx.x & 31;
  const int wave  = threadIdx.x >> 5;
  const int waveM = wave >> 2;
  const int waveN = wave & 3;
  const int l15   = lane & 15;
  const int half  = lane >> 4;
  const int mBlock = blockIdx.y * 128;
  const int nBlock = blockIdx.x * 128;

  v8f zero = {0.f, 0.f, 0.f, 0.f, 0.f, 0.f, 0.f, 0.f};
  v8f acc[4][2];
#pragma unroll
  for (int mt = 0; mt < 4; ++mt)
#pragma unroll
    for (int nt = 0; nt < 2; ++nt) acc[mt][nt] = zero;

  gemm_core(Ctx, Wo, As, Bs, mBlock, nBlock, acc);

#pragma unroll
  for (int mt = 0; mt < 4; ++mt)
#pragma unroll
    for (int nt = 0; nt < 2; ++nt)
#pragma unroll
      for (int j = 0; j < 8; ++j) {
        int m = mBlock + waveM * 64 + mt * 16 + j + 8 * half;
        int n = nBlock + waveN * 32 + nt * 16 + l15;
        Out[(size_t)m * E_DIM + n] = acc[mt][nt][j];
      }
}

// ---------------------------------------------------------------------------
// Launch
// ---------------------------------------------------------------------------
extern "C" void kernel_launch(void* const* d_in, const int* in_sizes, int n_in,
                              void* d_out, int out_size, void* d_ws, size_t ws_size,
                              hipStream_t stream) {
  const float* x  = (const float*)d_in[0];
  const float* Wq = (const float*)d_in[1];
  const float* Wk = (const float*)d_in[2];
  const float* Wv = (const float*)d_in[3];
  const float* Wo = (const float*)d_in[4];
  const float* qp = (const float*)d_in[5];

  const size_t MB = 1024 * 1024;
  char* ws = (char*)d_ws;
  _Float16* xh  = (_Float16*)(ws + 0);        // 8 MB (x), reused as ctx
  _Float16* ctx = xh;
  _Float16* wqh = (_Float16*)(ws + 8 * MB);
  _Float16* wkh = (_Float16*)(ws + 10 * MB);
  _Float16* wvh = (_Float16*)(ws + 12 * MB);
  _Float16* wvv = (_Float16*)(ws + 14 * MB);  // Wo
  _Float16* Qb  = (_Float16*)(ws + 16 * MB);  // head-major [B,H,S,dk]
  _Float16* Kb  = (_Float16*)(ws + 24 * MB);  // head-major [B,H,S,dk]
  _Float16* Vb  = (_Float16*)(ws + 32 * MB);  // d-major   [B,H,dk,S]

  const int nX4 = MTOT * E_DIM / 4;
  const int nW4 = E_DIM * E_DIM / 4;
  cast_f32_to_f16<<<1024, 256, 0, stream>>>(x,  xh,  nX4);
  cast_f32_to_f16<<<512,  256, 0, stream>>>(Wq, wqh, nW4);
  cast_f32_to_f16<<<512,  256, 0, stream>>>(Wk, wkh, nW4);
  cast_f32_to_f16<<<512,  256, 0, stream>>>(Wv, wvh, nW4);
  cast_f32_to_f16<<<512,  256, 0, stream>>>(Wo, wvv, nW4);

  qkv_gemm<<<dim3(E_DIM / 128, MTOT / 128, 3), 256, 0, stream>>>(
      xh, wqh, wkh, wvh, Qb, Kb, Vb);

  attn_kernel<<<dim3(SEQ / 64, BATCH * HEADS), 128, 0, stream>>>(
      Qb, Kb, Vb, qp, ctx);

  out_gemm<<<dim3(E_DIM / 128, MTOT / 128), 256, 0, stream>>>(
      ctx, wvv, (float*)d_out);
}
